// AdaptiveBatchPointnetSAModule_72705206386968
// MI455X (gfx1250) — matrix operand order repaired and checked by reading, not compiled, not run on hardware
//
#include <hip/hip_runtime.h>
#include <float.h>

typedef __attribute__((ext_vector_type(16))) _Float16 v16h;
typedef __attribute__((ext_vector_type(8)))  _Float16 v8h;
typedef __attribute__((ext_vector_type(8)))  float    v8f;

#define QN 16384
#define KN 64
#define CN 64

// LDS row strides (in halfs), multiples of 8 so rows stay 16B aligned
#define S0 104   // H0: 64 x 96 valid (cols: 0..63 feat, 64..66 xyz, 67..95 zero)
#define S1 72    // H1: 64 x 64
#define S2 136   // H2: 64 x 128

// workspace layout (half offsets for weights, byte offset for biases)
#define W1T_OFF 0                    // [64 ][96 ] f16
#define W2T_OFF 6144                 // [128][64 ] f16
#define W3T_OFF (6144 + 8192)        // [256][128] f16
#define WT_HALFS (6144 + 8192 + 32768)
#define BIAS_BYTE_OFF (WT_HALFS * 2) // then 448 f32 biases

// ---------------------------------------------------------------------------
// Fragment loaders matching the CDNA5 WMMA VGPR layouts (CDNA5 ISA)
// ---------------------------------------------------------------------------

// A (16x32 f16): lane L -> row M = L%16.
//   lanes 0-15 : halves 0..7 = K 0..7,  halves 8..15 = K 16..23
//   lanes 16-31: halves 0..7 = K 8..15, halves 8..15 = K 24..31
__device__ __forceinline__ v16h load_frag_A(const _Float16* H, int stride,
                                            int mtile, int kk, int lane) {
    int m  = mtile * 16 + (lane & 15);
    int b0 = (lane >> 4) * 8;
    const v8h* p0 = (const v8h*)(H + m * stride + kk * 32 + b0);
    const v8h* p1 = (const v8h*)(H + m * stride + kk * 32 + 16 + b0);
    v8h lo = *p0, hi = *p1;
    v16h r;
#pragma unroll
    for (int i = 0; i < 8; ++i) { r[i] = lo[i]; r[8 + i] = hi[i]; }
    return r;
}

// B (32x16 f16): lane L -> column N = L%16 (same K layout as A)
//   lanes 0-15 : halves 0..7 = K 0..7,  halves 8..15 = K 16..23
//   lanes 16-31: halves 0..7 = K 8..15, halves 8..15 = K 24..31
// Weights stored transposed [Nout][Kdim] so this is two 16B loads.
__device__ __forceinline__ v16h load_frag_B(const _Float16* __restrict__ Wt, int Kdim,
                                            int ntile, int kk, int lane) {
    int n  = ntile * 16 + (lane & 15);
    int k0 = kk * 32 + (lane >> 4) * 8;
    const v8h* p = (const v8h*)(Wt + n * Kdim + k0);
    v8h lo = p[0], hi = p[2];
    v16h r;
#pragma unroll
    for (int i = 0; i < 8; ++i) { r[i] = lo[i]; r[8 + i] = hi[i]; }
    return r;
}

// C/D (16x16 f32): VGPR r -> lanes 0-15: (M=r, N=lane); lanes 16-31: (M=r+8, N=lane-16)
__device__ __forceinline__ void store_frag_relu(v8f acc, _Float16* H, int stride,
                                                int mtile, int ntile, int lane,
                                                const float* __restrict__ bias) {
    int col   = ntile * 16 + (lane & 15);
    float bv  = bias[col];
    int rbase = mtile * 16 + ((lane >> 4) << 3);
#pragma unroll
    for (int r = 0; r < 8; ++r) {
        float v = acc[r] + bv;
        v = v > 0.f ? v : 0.f;
        H[(rbase + r) * stride + col] = (_Float16)v;
    }
}

__device__ __forceinline__ v8f wmma_f16(v16h a, v16h b, v8f c) {
    return __builtin_amdgcn_wmma_f32_16x16x32_f16(false, a, false, b, (short)0, c,
                                                  false, false);
}

// ---------------------------------------------------------------------------
// Prep: fold BN into weights, transpose to [Nout][Kdim] f16, build fused biases.
// Layer-1 K order matches H0: k<64 -> feature k (orig row k+3),
//                             64..66 -> xyz (orig rows 0..2), 67..95 -> 0.
// ---------------------------------------------------------------------------
__global__ void __launch_bounds__(256) prep_weights(
    const float* __restrict__ W1, const float* __restrict__ b1,
    const float* __restrict__ g1, const float* __restrict__ be1,
    const float* __restrict__ W2, const float* __restrict__ b2,
    const float* __restrict__ g2, const float* __restrict__ be2,
    const float* __restrict__ W3, const float* __restrict__ b3,
    const float* __restrict__ g3, const float* __restrict__ be3,
    _Float16* __restrict__ wt, float* __restrict__ bias) {
    int t = threadIdx.x + blockIdx.x * blockDim.x;
    int stride = blockDim.x * gridDim.x;
    // W1T: [64][96]
    for (int i = t; i < 64 * 96; i += stride) {
        int n = i / 96, k = i % 96;
        float v;
        if (k < 64)       v = W1[(k + 3) * 64 + n] * g1[n];  // feature rows
        else if (k < 67)  v = W1[(k - 64) * 64 + n] * g1[n]; // xyz rows
        else              v = 0.f;
        wt[W1T_OFF + i] = (_Float16)v;
    }
    // W2T: [128][64]
    for (int i = t; i < 128 * 64; i += stride) {
        int n = i / 64, k = i % 64;
        wt[W2T_OFF + i] = (_Float16)(W2[k * 128 + n] * g2[n]);
    }
    // W3T: [256][128]
    for (int i = t; i < 256 * 128; i += stride) {
        int n = i / 128, k = i % 128;
        wt[W3T_OFF + i] = (_Float16)(W3[k * 256 + n] * g3[n]);
    }
    if (t < 64)  bias[t]       = b1[t] * g1[t] + be1[t];
    if (t < 128) bias[64 + t]  = b2[t] * g2[t] + be2[t];
    if (t < 256) bias[192 + t] = b3[t] * g3[t] + be3[t];
}

// ---------------------------------------------------------------------------
// Main: one block (8 waves, 256 threads) per query.
//   gather -> H0(64x96) -> L1 -> H1(64x64) -> L2 -> H2(64x128) -> L3 fused max
// ---------------------------------------------------------------------------
__global__ void __launch_bounds__(256) sa_module_kernel(
    const float* __restrict__ xyz, const float* __restrict__ feat,
    const int* __restrict__ query_idx, const int* __restrict__ group_idx,
    const unsigned char* __restrict__ group_mask,
    const _Float16* __restrict__ wt, const float* __restrict__ bias,
    float* __restrict__ out) {
    __shared__ __align__(16) _Float16 sH0[64 * S0];
    __shared__ __align__(16) _Float16 sH1[64 * S1];
    __shared__ __align__(16) _Float16 sH2[64 * S2];
    __shared__ unsigned char smask[64];

    const int q    = blockIdx.x;
    const int tid  = threadIdx.x;
    const int wave = tid >> 5;
    const int lane = tid & 31;

    const _Float16* W1T = wt + W1T_OFF;
    const _Float16* W2T = wt + W2T_OFF;
    const _Float16* W3T = wt + W3T_OFF;
    const float* bias1 = bias;
    const float* bias2 = bias + 64;
    const float* bias3 = bias + 192;

    const int qi = query_idx[q];

    if (tid < 3) out[q * 3 + tid] = xyz[qi * 3 + tid];     // query_xyz output
    if (tid < 64) smask[tid] = group_mask[q * 64 + tid];

    // ---- gather phase (vectorized): features -> cols 0..63 of H0 ----
    {
        const int m    = tid >> 2;           // neighbor row 0..63
        const int jseg = tid & 3;            // 4 threads per row, 16 cols each
        const int gi   = group_idx[q * 64 + m];
        const float4* fp = (const float4*)(feat + gi * 64) + jseg * 4;
        float4 fa = fp[0], fb = fp[1], fc = fp[2], fd = fp[3];
        v8h h0, h1;
        h0[0] = (_Float16)fa.x; h0[1] = (_Float16)fa.y;
        h0[2] = (_Float16)fa.z; h0[3] = (_Float16)fa.w;
        h0[4] = (_Float16)fb.x; h0[5] = (_Float16)fb.y;
        h0[6] = (_Float16)fb.z; h0[7] = (_Float16)fb.w;
        h1[0] = (_Float16)fc.x; h1[1] = (_Float16)fc.y;
        h1[2] = (_Float16)fc.z; h1[3] = (_Float16)fc.w;
        h1[4] = (_Float16)fd.x; h1[5] = (_Float16)fd.y;
        h1[6] = (_Float16)fd.z; h1[7] = (_Float16)fd.w;
        _Float16* dst = sH0 + m * S0 + jseg * 16;
        *(v8h*)(dst)     = h0;
        *(v8h*)(dst + 8) = h1;
    }
    // ---- rel_xyz -> cols 64..66, zeros -> 67..95 (one row per thread) ----
    if (tid < 64) {
        const int m  = tid;
        const int gi = group_idx[q * 64 + m];
        const float qx = xyz[qi * 3 + 0];
        const float qy = xyz[qi * 3 + 1];
        const float qz = xyz[qi * 3 + 2];
        v8h hx = {};
        hx[0] = (_Float16)(xyz[gi * 3 + 0] - qx);
        hx[1] = (_Float16)(xyz[gi * 3 + 1] - qy);
        hx[2] = (_Float16)(xyz[gi * 3 + 2] - qz);
        v8h z = {};
        _Float16* row = sH0 + m * S0;
        *(v8h*)(row + 64) = hx;
        *(v8h*)(row + 72) = z;
        *(v8h*)(row + 80) = z;
        *(v8h*)(row + 88) = z;
    }
    __syncthreads();

    // ---- Layer 1: H1(64x64) = relu(H0(64x96) x W1T), 16 tiles / 8 waves ----
    {
        const int mt    = wave & 3;
        const int npair = wave >> 2;
#pragma unroll
        for (int j = 0; j < 2; ++j) {
            const int nt = npair * 2 + j;
            v8f acc = {};
#pragma unroll
            for (int kk = 0; kk < 3; ++kk) {
                v16h a = load_frag_A(sH0, S0, mt, kk, lane);
                v16h b = load_frag_B(W1T, 96, nt, kk, lane);
                acc = wmma_f16(a, b, acc);
            }
            store_frag_relu(acc, sH1, S1, mt, nt, lane, bias1);
        }
    }
    __syncthreads();

    // ---- Layer 2: H2(64x128) = relu(H1 x W2T), wave owns one N-tile ----
    {
        const int nt = wave;
        v8f acc0 = {}, acc1 = {}, acc2 = {}, acc3 = {};
#pragma unroll
        for (int kk = 0; kk < 2; ++kk) {
            v16h b = load_frag_B(W2T, 64, nt, kk, lane);
            acc0 = wmma_f16(load_frag_A(sH1, S1, 0, kk, lane), b, acc0);
            acc1 = wmma_f16(load_frag_A(sH1, S1, 1, kk, lane), b, acc1);
            acc2 = wmma_f16(load_frag_A(sH1, S1, 2, kk, lane), b, acc2);
            acc3 = wmma_f16(load_frag_A(sH1, S1, 3, kk, lane), b, acc3);
        }
        store_frag_relu(acc0, sH2, S2, 0, nt, lane, bias2);
        store_frag_relu(acc1, sH2, S2, 1, nt, lane, bias2);
        store_frag_relu(acc2, sH2, S2, 2, nt, lane, bias2);
        store_frag_relu(acc3, sH2, S2, 3, nt, lane, bias2);
    }
    __syncthreads();

    // ---- Layer 3 fused: relu(H2 x W3T) + mask + max over 64 neighbors ----
    {
        const int halfsel = (lane >> 4) << 3;   // 0 or 8: which row-half lane holds
#pragma unroll
        for (int j = 0; j < 2; ++j) {
            const int nt = wave * 2 + j;
            v8f acc0 = {}, acc1 = {}, acc2 = {}, acc3 = {};
#pragma unroll
            for (int kk = 0; kk < 4; ++kk) {
                v16h b = load_frag_B(W3T, 128, nt, kk, lane);
                acc0 = wmma_f16(load_frag_A(sH2, S2, 0, kk, lane), b, acc0);
                acc1 = wmma_f16(load_frag_A(sH2, S2, 1, kk, lane), b, acc1);
                acc2 = wmma_f16(load_frag_A(sH2, S2, 2, kk, lane), b, acc2);
                acc3 = wmma_f16(load_frag_A(sH2, S2, 3, kk, lane), b, acc3);
            }
            const float bv = bias3[nt * 16 + (lane & 15)];
            float mx = -FLT_MAX;
#pragma unroll
            for (int r = 0; r < 8; ++r) {
                int row = r + halfsel;  // rows within each 16-row M tile
                float v0 = acc0[r] + bv; v0 = v0 > 0.f ? v0 : 0.f;
                float v1 = acc1[r] + bv; v1 = v1 > 0.f ? v1 : 0.f;
                float v2 = acc2[r] + bv; v2 = v2 > 0.f ? v2 : 0.f;
                float v3 = acc3[r] + bv; v3 = v3 > 0.f ? v3 : 0.f;
                if (!smask[row])      v0 = -FLT_MAX;
                if (!smask[16 + row]) v1 = -FLT_MAX;
                if (!smask[32 + row]) v2 = -FLT_MAX;
                if (!smask[48 + row]) v3 = -FLT_MAX;
                mx = fmaxf(mx, fmaxf(fmaxf(v0, v1), fmaxf(v2, v3)));
            }
            // combine the two row-halves (lane L <-> L^16 hold same column)
            mx = fmaxf(mx, __shfl_xor(mx, 16, 32));
            if (lane < 16)
                out[QN * 3 + q * 256 + nt * 16 + lane] = mx;
        }
    }
}

// ---------------------------------------------------------------------------
extern "C" void kernel_launch(void* const* d_in, const int* in_sizes, int n_in,
                              void* d_out, int out_size, void* d_ws, size_t ws_size,
                              hipStream_t stream) {
    const float* xyz  = (const float*)d_in[0];
    const float* feat = (const float*)d_in[1];
    const int* query_idx = (const int*)d_in[2];
    const int* group_idx = (const int*)d_in[3];
    const unsigned char* group_mask = (const unsigned char*)d_in[4];
    const float* W1  = (const float*)d_in[5];
    const float* b1  = (const float*)d_in[6];
    const float* g1  = (const float*)d_in[7];
    const float* be1 = (const float*)d_in[8];
    const float* W2  = (const float*)d_in[9];
    const float* b2  = (const float*)d_in[10];
    const float* g2  = (const float*)d_in[11];
    const float* be2 = (const float*)d_in[12];
    const float* W3  = (const float*)d_in[13];
    const float* b3  = (const float*)d_in[14];
    const float* g3  = (const float*)d_in[15];
    const float* be3 = (const float*)d_in[16];

    _Float16* wt = (_Float16*)d_ws;
    float* bias  = (float*)((char*)d_ws + BIAS_BYTE_OFF);
    float* out   = (float*)d_out;

    prep_weights<<<64, 256, 0, stream>>>(W1, b1, g1, be1, W2, b2, g2, be2,
                                         W3, b3, g3, be3, wt, bias);
    sa_module_kernel<<<QN, 256, 0, stream>>>(xyz, feat, query_idx, group_idx,
                                             group_mask, wt, bias, out);
}
